// MultiHeadAttention_30004641530452
// MI455X (gfx1250) — matrix operand order repaired and checked by reading, not compiled, not run on hardware
//
#include <hip/hip_runtime.h>
#include <hip/hip_bf16.h>
#include <cstdint>
#include <cstddef>

#define DEVINL __device__ __forceinline__

typedef __attribute__((ext_vector_type(16))) __bf16 v16bf;
typedef __attribute__((ext_vector_type(8)))  __bf16 v8bf;
typedef __attribute__((ext_vector_type(8)))  float  v8f;
typedef __attribute__((ext_vector_type(4)))  unsigned int v4u;
typedef __attribute__((ext_vector_type(8)))  unsigned int v8u;

constexpr int Bc  = 4;
constexpr int Sc  = 2048;
constexpr int Dc  = 1024;
constexpr int Hc  = 16;
constexpr int DKc = 64;
constexpr int NT  = Bc * Sc;   // 8192 tokens
constexpr int KCH = 256;       // K-chunk staged in LDS per TDM transfer

// ---------------------------------------------------------------------------
// A-operand fragment: 16x32 bf16, row-major source [M, K] stride ld.
// Lanes 0-15: M=lane, K = k0+{0..7} (v0-3), k0+16+{0..7} (v4-7).
// Lanes 16-31: M=lane-16, K = k0+8+{0..7}, k0+24+{0..7}.
DEVINL v16bf load_frag_a(const __bf16* __restrict__ p, int ld, int row0, int k0) {
  const int lane = threadIdx.x & 31;
  const int half = lane >> 4;
  const int m    = lane & 15;
  const __bf16* base = p + (size_t)(row0 + m) * ld + (k0 + half * 8);
  v8bf lo = *(const v8bf*)(base);
  v8bf hi = *(const v8bf*)(base + 16);
  v16bf r;
#pragma unroll
  for (int i = 0; i < 8; ++i) { r[i] = lo[i]; r[i + 8] = hi[i]; }
  return r;
}

// B-operand fragment: B is 32x16 (KxN); source is B^T row-major [N, K].
// Lanes 0-15: N=lane, K=k0+{0..7},k0+16+{0..7}; lanes 16-31: N=lane-16, K=k0+8+{0..7},k0+24+{0..7}.
DEVINL v16bf load_frag_b(const __bf16* __restrict__ p, int ld, int n0, int k0) {
  const int lane = threadIdx.x & 31;
  const int half = lane >> 4;
  const int n    = lane & 15;
  const __bf16* base = p + (size_t)(n0 + n) * ld + (k0 + half * 8);
  v8bf lo = *(const v8bf*)(base);
  v8bf hi = *(const v8bf*)(base + 16);
  v16bf r;
#pragma unroll
  for (int i = 0; i < 8; ++i) { r[i] = lo[i]; r[i + 8] = hi[i]; }
  return r;
}

DEVINL v8f wmma_bf16(v16bf a, v16bf b, v8f c) {
  return __builtin_amdgcn_wmma_f32_16x16x32_bf16(false, a, false, b, (short)0, c,
                                                 false, false);
}

// TDM: DMA a [rows=64, cols=KCH] bf16 tile (row-major, row stride ld elements)
// from global into LDS (packed rows of KCH). 2D descriptor, groups 0+1 only.
DEVINL void tdm_load_tile_2d(const __bf16* gsrc, unsigned lds_byte_off, int ld) {
  const unsigned long long ga = (unsigned long long)(uintptr_t)gsrc;
  v4u g0;
  g0[0] = 1u;                                   // count=1, user descriptor
  g0[1] = lds_byte_off;                         // lds_addr
  g0[2] = (unsigned)(ga & 0xffffffffu);         // global_addr[31:0]
  g0[3] = (unsigned)((ga >> 32) & 0x1ffffffu) | (2u << 30);  // addr[56:32], type=2
  v8u g1;
  g1[0] = 0x00010000u;                          // wg_mask=0, data_size=1 (2 bytes)
  g1[1] = ((unsigned)ld & 0xffffu) << 16;       // tensor_dim0[15:0]
  g1[2] = (((unsigned)ld >> 16) & 0xffffu) |    // tensor_dim0[31:16]
          (((unsigned)ld & 0xffffu) << 16);     // tensor_dim1[15:0]
  g1[3] = (((unsigned)ld >> 16) & 0xffffu) |    // tensor_dim1[31:16]
          ((unsigned)KCH << 16);                // tile_dim0 = KCH
  g1[4] = 64u;                                  // tile_dim1 = 64 rows, tile_dim2=0
  g1[5] = (unsigned)ld;                         // tensor_dim0_stride[31:0]
  g1[6] = 0u;
  g1[7] = 0u;
  asm volatile("tensor_load_to_lds %0, %1" :: "s"(g0), "s"(g1) : "memory");
}

// ---------------------------------------------------------------------------
__global__ void f32_to_bf16_kernel(const float* __restrict__ in,
                                   __bf16* __restrict__ out, int n) {
  int i = blockIdx.x * blockDim.x + threadIdx.x;
  const int stride = gridDim.x * blockDim.x;
  for (; i < n; i += stride) out[i] = (__bf16)in[i];
}

// ---------------------------------------------------------------------------
// Y[m, n] = sum_k X[m,k] * W[n,k] + bias[n]
// Block: 4 waves share a 64-column W panel; the TDM streams it into LDS with
// double buffering (issue chunk c+1, s_wait_tensorcnt(1) drains only chunk c,
// compute overlaps the DMA). Each wave: 16-row M tile x four 16-col N tiles.
// mode 0: bf16 [M, Dc];  mode 1: bf16 [B,H,S,DK];  mode 2: bf16 [B,H,DK,S];
// mode 3: fp32 [M, Dc].
__global__ void __launch_bounds__(128)
gemm_bf16_wmma(const __bf16* __restrict__ X, const __bf16* __restrict__ W,
               const float* __restrict__ bias, __bf16* __restrict__ outb,
               float* __restrict__ outf, int mode) {
  __shared__ __bf16 wpanel[2][64 * KCH];  // 2 x 32 KB, double-buffered

  const int tM   = blockIdx.y * 4 + threadIdx.y;  // 0..511
  const int n0   = blockIdx.x * 64;               // panel base column
  const int row0 = tM * 16;
  constexpr int NCHUNK = Dc / KCH;                // 4

  if (threadIdx.y == 0) {
    tdm_load_tile_2d(W + (size_t)n0 * Dc,
                     (unsigned)(uintptr_t)(&wpanel[0][0]), Dc);
  }

  v8f c[4] = {};

  for (int ch = 0; ch < NCHUNK; ++ch) {
    if (threadIdx.y == 0) {
      if (ch + 1 < NCHUNK) {
        tdm_load_tile_2d(W + (size_t)n0 * Dc + (ch + 1) * KCH,
                         (unsigned)(uintptr_t)(&wpanel[(ch + 1) & 1][0]), Dc);
        __builtin_amdgcn_s_wait_tensorcnt(1);  // chunk ch done (in-order TDM)
      } else {
        __builtin_amdgcn_s_wait_tensorcnt(0);
      }
    }
    __syncthreads();  // publish chunk ch to all waves

    const __bf16* Wl = &wpanel[ch & 1][0];
    const int k0 = ch * KCH;
#pragma unroll
    for (int kk = 0; kk < KCH; kk += 32) {
      v16bf a  = load_frag_a(X, Dc, row0, k0 + kk);
      v16bf b0 = load_frag_b(Wl, KCH, 0,  kk);
      v16bf b1 = load_frag_b(Wl, KCH, 16, kk);
      v16bf b2 = load_frag_b(Wl, KCH, 32, kk);
      v16bf b3 = load_frag_b(Wl, KCH, 48, kk);
      c[0] = wmma_bf16(a, b0, c[0]);
      c[1] = wmma_bf16(a, b1, c[1]);
      c[2] = wmma_bf16(a, b2, c[2]);
      c[3] = wmma_bf16(a, b3, c[3]);
    }
    __syncthreads();  // all waves done with buffer (ch&1) before it is refilled
  }

  const int lane = threadIdx.x & 31;
  const int half = lane >> 4;
  const int nl   = lane & 15;
#pragma unroll
  for (int j = 0; j < 4; ++j) {
    const int ncol = n0 + j * 16 + nl;
    const float bv = bias[ncol];
#pragma unroll
    for (int r = 0; r < 8; ++r) {
      const int mrow = row0 + r + half * 8;
      const float val = c[j][r] + bv;
      if (mode == 3) {
        outf[(size_t)mrow * Dc + ncol] = val;
      } else if (mode == 0) {
        outb[(size_t)mrow * Dc + ncol] = (__bf16)val;
      } else {
        const int bb = mrow >> 11;          // / Sc
        const int ss = mrow & (Sc - 1);
        const int hh = ncol >> 6;           // / DKc
        const int dk = ncol & (DKc - 1);
        if (mode == 1)
          outb[(((size_t)(bb * Hc + hh)) * Sc + ss) * DKc + dk] = (__bf16)val;
        else
          outb[(((size_t)(bb * Hc + hh)) * DKc + dk) * Sc + ss] = (__bf16)val;
      }
    }
  }
}

// ---------------------------------------------------------------------------
// Flash-attention: Q,K in [B,H,S,DK] bf16; VT in [B,H,DK,S] bf16.
// Each wave: 16 query rows, streaming 32-key blocks with online softmax.
__global__ void __launch_bounds__(128)
attn_wmma(const __bf16* __restrict__ Q, const __bf16* __restrict__ K,
          const __bf16* __restrict__ VT, __bf16* __restrict__ CTX) {
  __shared__ __bf16 lds_p[4][16][32];   // per-wave P staging (C-layout -> A-layout)

  const int wave = threadIdx.y;
  const int lane = threadIdx.x & 31;
  const int half = lane >> 4;
  const int nl   = lane & 15;
  const int h    = blockIdx.y;
  const int b    = blockIdx.z;
  const int q0   = (blockIdx.x * 4 + wave) * 16;

  const __bf16* Qh = Q  + ((size_t)(b * Hc + h)) * Sc * DKc;  // [S, DK]
  const __bf16* Kh = K  + ((size_t)(b * Hc + h)) * Sc * DKc;  // [S, DK]
  const __bf16* Vh = VT + ((size_t)(b * Hc + h)) * DKc * Sc;  // [DK, S]

  const v16bf qa0 = load_frag_a(Qh, DKc, q0, 0);
  const v16bf qa1 = load_frag_a(Qh, DKc, q0, 32);

  v8f o[4] = {};
  float rmax[8], rsum[8];
#pragma unroll
  for (int r = 0; r < 8; ++r) { rmax[r] = -1e30f; rsum[r] = 0.0f; }

  const float scale = 0.125f;  // 1/sqrt(DK)

  for (int kb = 0; kb < Sc; kb += 32) {
    // V fragments for this key block: issue early, consume after softmax.
    v16bf vf0 = load_frag_b(Vh, Sc, 0,  kb);
    v16bf vf1 = load_frag_b(Vh, Sc, 16, kb);
    v16bf vf2 = load_frag_b(Vh, Sc, 32, kb);
    v16bf vf3 = load_frag_b(Vh, Sc, 48, kb);

    // scores for 32 keys: two 16-key C tiles
    v8f s0 = {}, s1 = {};
    {
      v16bf k00 = load_frag_b(Kh, DKc, kb, 0);
      v16bf k01 = load_frag_b(Kh, DKc, kb, 32);
      v16bf k10 = load_frag_b(Kh, DKc, kb + 16, 0);
      v16bf k11 = load_frag_b(Kh, DKc, kb + 16, 32);
      s0 = wmma_bf16(qa0, k00, s0);
      s0 = wmma_bf16(qa1, k01, s0);
      s1 = wmma_bf16(qa0, k10, s1);
      s1 = wmma_bf16(qa1, k11, s1);
    }

    // online softmax per row (register r holds rows r / r+8 across lane halves)
#pragma unroll
    for (int r = 0; r < 8; ++r) {
      const float v0 = s0[r] * scale;
      const float v1 = s1[r] * scale;
      float tmax = fmaxf(v0, v1);
#pragma unroll
      for (int off = 1; off < 16; off <<= 1)
        tmax = fmaxf(tmax, __shfl_xor(tmax, off, 32));
      const float nm = fmaxf(rmax[r], tmax);
      const float p0 = __expf(v0 - nm);
      const float p1 = __expf(v1 - nm);
      float ps = p0 + p1;
#pragma unroll
      for (int off = 1; off < 16; off <<= 1) ps += __shfl_xor(ps, off, 32);
      const float corr = __expf(rmax[r] - nm);
      rsum[r] = rsum[r] * corr + ps;
      rmax[r] = nm;
#pragma unroll
      for (int t = 0; t < 4; ++t) o[t][r] *= corr;

      const int row = r + half * 8;
      lds_p[wave][row][nl]      = (__bf16)p0;
      lds_p[wave][row][16 + nl] = (__bf16)p1;
    }

    // P (16x32) back in A-layout; accumulate ctx += P @ V_block
    const v16bf pf = load_frag_a(&lds_p[wave][0][0], 32, 0, 0);
    o[0] = wmma_bf16(pf, vf0, o[0]);
    o[1] = wmma_bf16(pf, vf1, o[1]);
    o[2] = wmma_bf16(pf, vf2, o[2]);
    o[3] = wmma_bf16(pf, vf3, o[3]);
  }

  // finalize: divide by softmax denom, store ctx as [token, h*DK + dk]
#pragma unroll
  for (int t = 0; t < 4; ++t) {
    const int feature = h * DKc + t * 16 + nl;
#pragma unroll
    for (int r = 0; r < 8; ++r) {
      const int m     = r + half * 8;
      const int token = b * Sc + q0 + m;
      CTX[(size_t)token * Dc + feature] = (__bf16)(o[t][r] / rsum[r]);
    }
  }
}

// ---------------------------------------------------------------------------
extern "C" void kernel_launch(void* const* d_in, const int* in_sizes, int n_in,
                              void* d_out, int out_size, void* d_ws, size_t ws_size,
                              hipStream_t stream) {
  (void)in_sizes; (void)n_in; (void)out_size; (void)ws_size;
  const float* q_in = (const float*)d_in[0];
  const float* k_in = (const float*)d_in[1];
  const float* v_in = (const float*)d_in[2];
  const float* Wq = (const float*)d_in[3];
  const float* bq = (const float*)d_in[4];
  const float* Wk = (const float*)d_in[5];
  const float* bk = (const float*)d_in[6];
  const float* Wv = (const float*)d_in[7];
  const float* bv = (const float*)d_in[8];
  const float* Wo = (const float*)d_in[9];
  const float* bo = (const float*)d_in[10];
  float* out = (float*)d_out;

  char* ws = (char*)d_ws;
  size_t off = 0;
  auto alloc = [&](size_t elems) -> __bf16* {
    __bf16* p = (__bf16*)(ws + off);
    off = (off + elems * sizeof(__bf16) + 255) & ~(size_t)255;
    return p;
  };
  const size_t ND = (size_t)NT * Dc;
  const size_t DD = (size_t)Dc * Dc;
  __bf16* Xq  = alloc(ND);
  __bf16* Xk  = alloc(ND);
  __bf16* Xv  = alloc(ND);
  __bf16* Wqb = alloc(DD);
  __bf16* Wkb = alloc(DD);
  __bf16* Wvb = alloc(DD);
  __bf16* Wob = alloc(DD);
  __bf16* Qb  = alloc(ND);
  __bf16* Kb  = alloc(ND);
  __bf16* VTb = alloc(ND);
  __bf16* CTX = alloc(ND);

  auto conv = [&](const float* src, __bf16* dst, size_t n) {
    int blocks = (int)((n + 255) / 256);
    if (blocks > 4096) blocks = 4096;
    f32_to_bf16_kernel<<<dim3(blocks), dim3(256), 0, stream>>>(src, dst, (int)n);
  };
  conv(q_in, Xq, ND);
  conv(k_in, Xk, ND);
  conv(v_in, Xv, ND);
  conv(Wq, Wqb, DD);
  conv(Wk, Wkb, DD);
  conv(Wv, Wvb, DD);
  conv(Wo, Wob, DD);

  const dim3 gblk(32, 4);
  const dim3 ggrid(Dc / 64, NT / 64);  // (16, 128)
  gemm_bf16_wmma<<<ggrid, gblk, 0, stream>>>(Xq, Wqb, bq, Qb, nullptr, 1);
  gemm_bf16_wmma<<<ggrid, gblk, 0, stream>>>(Xk, Wkb, bk, Kb, nullptr, 1);
  gemm_bf16_wmma<<<ggrid, gblk, 0, stream>>>(Xv, Wvb, bv, VTb, nullptr, 2);

  const dim3 ablk(32, 4);
  const dim3 agrid(Sc / 64, Hc, Bc);   // (32, 16, 4)
  attn_wmma<<<agrid, ablk, 0, stream>>>(Qb, Kb, VTb, CTX);

  gemm_bf16_wmma<<<ggrid, gblk, 0, stream>>>(CTX, Wob, bo, nullptr, out, 3);
}
